// CTC_54460185313886
// MI455X (gfx1250) — compile-verified, matrix operand order checked by reading, not run on hardware
//
#include <hip/hip_runtime.h>

typedef __bf16 v16bf __attribute__((ext_vector_type(16)));
typedef __bf16 v8bf  __attribute__((ext_vector_type(8)));
typedef float  v8f   __attribute__((ext_vector_type(8)));

#define BB 32
#define TT 800
#define DD 512
#define VV 1024
#define LL 100
#define SS 201            // 2*LL + 1
#define NEGV -1e30f

__device__ __forceinline__ __bf16 cvt_bf16(float f) {
  unsigned u = __builtin_bit_cast(unsigned, f);
  u += 0x7FFFu + ((u >> 16) & 1u);          // round-to-nearest-even
  unsigned short h = (unsigned short)(u >> 16);
  return __builtin_bit_cast(__bf16, h);
}

__device__ __forceinline__ float lae(float a, float b) {
  float m = fmaxf(a, b);
  float n = fminf(a, b);
  return m + log1pf(__expf(n - m));         // n - m <= 0
}

// ---------------------------------------------------------------------------
// Kernel W: one-shot f32 -> bf16 conversion of W into workspace (1 MB, reused
// by all GEMM blocks; stays resident in L2/WGP$).
// ---------------------------------------------------------------------------
__global__ __launch_bounds__(256)
void wcvt_kernel(const float* __restrict__ W, __bf16* __restrict__ Wb) {
  int i = blockIdx.x * 256 + threadIdx.x;   // grid sized exactly to V*D
  Wb[i] = cvt_bf16(W[i]);
}

// ---------------------------------------------------------------------------
// Kernel A: fused logits GEMM (bf16 WMMA) + log-softmax + ext-label gather.
// One block = 16 consecutive (b,t) rows. Per wave: k-outer loop with 8 live
// 16x16 accumulators; A-fragment loaded once per k-step from LDS and fed to
// 8 WMMAs. B-fragments are raw b128 loads of pre-converted bf16 W.
// ---------------------------------------------------------------------------
__global__ __launch_bounds__(256)
void ctc_logits_kernel(const float*  __restrict__ hpad,
                       const __bf16* __restrict__ Wb,
                       const float*  __restrict__ bias,
                       const int*    __restrict__ labels,
                       float*        __restrict__ lp) {
  extern __shared__ char smem[];
  __bf16 (*Albs)[DD]    = (__bf16 (*)[DD])(smem);                     // 16*512*2  = 16384 B
  float  (*Ltile)[1028] = (float  (*)[1028])(smem + 16384);           // 16*1028*4 = 65792 B
  float  (*redbuf)[16]  = (float  (*)[16])(smem + 16384 + 65792);     // 1024 B
  float* rowmax = (float*)(smem + 16384 + 65792 + 1024);              // 64 B
  float* rowlse = rowmax + 16;                                        // 64 B

  const int tid     = threadIdx.x;
  const int lane    = tid & 31;           // wave32
  const int wave    = tid >> 5;           // 8 waves / block
  const int rowbase = blockIdx.x * 16;    // flattened (b*T + t) row
  const int bidx    = rowbase / TT;

  // Stage A tile (16 rows x 512 K) into LDS as bf16.
  for (int idx = tid; idx < 16 * DD; idx += 256) {
    int m = idx >> 9, d = idx & (DD - 1);
    Albs[m][d] = cvt_bf16(hpad[(size_t)(rowbase + m) * DD + d]);
  }
  __syncthreads();

  const int nlo  = lane & 15;             // M for A-frag, N for B-frag
  const int koff = (lane >> 4) << 3;      // 0 (lanes 0-15) or 8 (lanes 16-31)
  const int mhi  = (lane >> 4) << 3;      // C/D row offset for this half-wave
  const int colbase0 = wave * 128;        // this wave owns cols [colbase0, +128)

  v8f acc[8] = {};
  // Per-lane base into bf16 W for this lane's N column of tile ci=0.
  const __bf16* wlane = Wb + (size_t)(colbase0 + nlo) * DD + koff;

#pragma unroll 2
  for (int kb = 0; kb < DD; kb += 32) {
    const v8bf* ap = (const v8bf*)&Albs[nlo][kb + koff];   // 16B-aligned
    v16bf afrag = __builtin_shufflevector(ap[0], ap[2],
        0, 1, 2, 3, 4, 5, 6, 7, 8, 9, 10, 11, 12, 13, 14, 15);
#pragma unroll
    for (int ci = 0; ci < 8; ++ci) {
      const v8bf* bp = (const v8bf*)(wlane + (size_t)ci * 16 * DD + kb);
      v16bf bfrag = __builtin_shufflevector(bp[0], bp[2],
          0, 1, 2, 3, 4, 5, 6, 7, 8, 9, 10, 11, 12, 13, 14, 15);
      acc[ci] = __builtin_amdgcn_wmma_f32_16x16x32_bf16(
          /*neg_a=*/false, afrag, /*neg_b=*/false, bfrag,
          /*c_mod=*/(short)0, acc[ci], /*reuse_a=*/false, /*reuse_b=*/false);
    }
  }

  // Writeback: VGPR i of acc[ci] -> (M = mhi+i, N = colbase0 + ci*16 + nlo).
#pragma unroll
  for (int ci = 0; ci < 8; ++ci) {
    const int ncol = colbase0 + ci * 16 + nlo;
    const float bv = bias[ncol];
#pragma unroll
    for (int i = 0; i < 8; ++i)
      Ltile[mhi + i][ncol] = acc[ci][i] + bv;
  }
  __syncthreads();

  // Row max over V (16 threads per row).
  {
    int m = tid >> 4, j = tid & 15;
    float mx = -3.4e38f;
    for (int c = j; c < VV; c += 16) mx = fmaxf(mx, Ltile[m][c]);
    redbuf[m][j] = mx;
  }
  __syncthreads();
  if (tid < 16) {
    float mx = redbuf[tid][0];
    for (int j = 1; j < 16; ++j) mx = fmaxf(mx, redbuf[tid][j]);
    rowmax[tid] = mx;
  }
  __syncthreads();
  // Row sum(exp(x - max)).
  {
    int m = tid >> 4, j = tid & 15;
    float mx = rowmax[m], s = 0.f;
    for (int c = j; c < VV; c += 16) s += __expf(Ltile[m][c] - mx);
    redbuf[m][j] = s;
  }
  __syncthreads();
  if (tid < 16) {
    float s = 0.f;
    for (int j = 0; j < 16; ++j) s += redbuf[tid][j];
    rowlse[tid] = rowmax[tid] + __logf(s);
  }
  __syncthreads();

  // Gather log-softmax at the 201 extended-label ids per row.
  const int* labrow = labels + bidx * LL;
  for (int idx = tid; idx < 16 * SS; idx += 256) {
    int m = idx / SS, s = idx - m * SS;
    int v = (s & 1) ? labrow[s >> 1] : 0;              // blank = 0
    lp[(size_t)(rowbase + m) * SS + s] = Ltile[m][v] - rowlse[m];
  }
}

// ---------------------------------------------------------------------------
// Kernel B: CTC forward recursion. One block per batch element; alpha double-
// buffered in LDS (states padded by 2 leading NEG cells so s-1/s-2 are
// branch-free). One barrier per frame.
// ---------------------------------------------------------------------------
__global__ __launch_bounds__(256)
void ctc_forward_kernel(const float* __restrict__ lp,
                        const int*   __restrict__ labels,
                        const int*   __restrict__ ilens,
                        const int*   __restrict__ label_lens,
                        float*       __restrict__ costs) {
  __shared__ float abuf[2][SS + 4];
  const int b   = blockIdx.x;
  const int tid = threadIdx.x;
  const float* lpb = lp + (size_t)b * TT * SS;

  bool allow_s = false;
  if (tid < SS && tid >= 3 && (tid & 1)) {
    int e  = labels[b * LL + (tid >> 1)];
    int e2 = labels[b * LL + (tid >> 1) - 1];
    allow_s = (e != e2);                     // skip transition s-2 -> s
  }
  const int lablen = label_lens[b];
  int Tlim = ilens[b];
  if (Tlim > TT) Tlim = TT;
  if (Tlim < 1)  Tlim = 1;

  if (tid < 2) { abuf[0][tid] = NEGV; abuf[1][tid] = NEGV; }
  if (tid < SS) {
    float a0 = NEGV;
    if (tid == 0)                    a0 = lpb[0];
    else if (tid == 1 && lablen > 0) a0 = lpb[1];
    abuf[0][tid + 2] = a0;
  }
  __syncthreads();

  int pc = 0;
  for (int t = 1; t < Tlim; ++t) {
    if (tid < SS) {
      float cur = abuf[pc][tid + 2];
      float p1  = abuf[pc][tid + 1];
      float p2  = abuf[pc][tid];
      float c = lae(cur, p1);
      if (allow_s) c = lae(c, p2);
      abuf[pc ^ 1][tid + 2] = c + lpb[(size_t)t * SS + tid];
    }
    __syncthreads();
    pc ^= 1;
  }

  if (tid == 0) {
    int il = 2 * lablen;
    if (il > SS - 1) il = SS - 1;
    float last = abuf[pc][il + 2];
    float prev = (lablen > 0) ? abuf[pc][il + 1] : NEGV;
    costs[b] = -lae(last, prev);
  }
}

// ---------------------------------------------------------------------------
// Kernel C: sum(costs)/B -> scalar output.
// ---------------------------------------------------------------------------
__global__ void ctc_reduce_kernel(const float* __restrict__ costs,
                                  float* __restrict__ out) {
  if (threadIdx.x == 0) {
    float s = 0.f;
    for (int i = 0; i < BB; ++i) s += costs[i];
    out[0] = s / (float)BB;
  }
}

extern "C" void kernel_launch(void* const* d_in, const int* in_sizes, int n_in,
                              void* d_out, int out_size, void* d_ws, size_t ws_size,
                              hipStream_t stream) {
  const float* hpad   = (const float*)d_in[0];
  const float* W      = (const float*)d_in[1];
  const float* bias   = (const float*)d_in[2];
  const int*   labels = (const int*)d_in[3];
  const int*   ilens  = (const int*)d_in[4];
  const int*   llens  = (const int*)d_in[5];

  float*  lp    = (float*)d_ws;                              // [B*T, S] = 20.6 MB
  float*  costs = lp + (size_t)BB * TT * SS;                 // [B]
  __bf16* Wb    = (__bf16*)((char*)d_ws +
                  ((size_t)BB * TT * SS + BB) * sizeof(float)); // 1 MB bf16 W

  const size_t smem = 16384 + 65792 + 1024 + 128;            // 83328 B dynamic LDS

  hipLaunchKernelGGL(wcvt_kernel, dim3(VV * DD / 256), dim3(256), 0, stream, W, Wb);
  hipLaunchKernelGGL(ctc_logits_kernel, dim3(BB * TT / 16), dim3(256), smem, stream,
                     hpad, Wb, bias, labels, lp);
  hipLaunchKernelGGL(ctc_forward_kernel, dim3(BB), dim3(256), 0, stream,
                     lp, labels, ilens, llens, costs);
  hipLaunchKernelGGL(ctc_reduce_kernel, dim3(1), dim3(32), 0, stream,
                     costs, (float*)d_out);
}
